// QLSTMQuantum_65481071404124
// MI455X (gfx1250) — matrix operand order repaired and checked by reading, not compiled, not run on hardware
//
#include <hip/hip_runtime.h>

// QLSTM on MI455X:
//  - precompute x-projections for all timesteps as one big parallel bf16 WMMA
//    GEMM with 4Mx2N register blocking (8 wmma per 12 fragment loads)
//  - per timestep: recurrent bf16 WMMA GEMM with h staged into LDS once per
//    workgroup via the Tensor Data Mover (tensor_load_to_lds + s_wait_tensorcnt),
//    weights streaming from the 192MB L2; then a fused pointwise cell update.

#define TSEQ  256
#define BATCH 64
#define DIN   1024
#define HID   1024
#define KCOMB (DIN + HID)       // 2048
#define ROWS  (TSEQ * BATCH)    // 16384
#define BH    (BATCH * HID)     // 65536

typedef __attribute__((ext_vector_type(16))) __bf16 v16bf;
typedef __attribute__((ext_vector_type(8)))  float  v8f;
typedef __attribute__((ext_vector_type(4)))  unsigned int v4u;
typedef __attribute__((ext_vector_type(4)))  int    v4i;
typedef __attribute__((ext_vector_type(8)))  int    v8i;

union FragB16 { v16bf v; uint4 q[2]; };

__device__ __forceinline__ unsigned short f2bf(float f) {
  unsigned int u = __float_as_uint(f);
  u += 0x7FFFu + ((u >> 16) & 1u);          // round-to-nearest-even
  return (unsigned short)(u >> 16);
}
__device__ __forceinline__ float bf2f(unsigned short h) {
  return __uint_as_float(((unsigned int)h) << 16);
}
__device__ __forceinline__ float sigmoidf_(float x) {
  return 1.0f / (1.0f + __expf(-x));
}

// --------------------------------------------------------------- TDM load ---
// 2-D tensor (dim1 rows x dim0 elems, 2-byte elements) -> LDS, whole tile.
// D# layout per CDNA5 ISA 8.3/8.4: group0 = {flags, lds_addr, global_addr,
// type=2}; group1 = {data_size, dims, tile dims, dim0 stride}.
__device__ __forceinline__ void tdm_load_2d_bf16_to_lds(
    const void* gaddr, unsigned ldsAddr, unsigned dim0, unsigned dim1,
    unsigned stride0) {
  unsigned long long ga = (unsigned long long)(uintptr_t)gaddr;
  v4u g0;
  g0[0] = 1u;                                    // count=1 valid descriptor
  g0[1] = ldsAddr;                               // lds_addr (bytes)
  g0[2] = (unsigned)(ga & 0xFFFFFFFFull);        // global_addr[31:0]
  g0[3] = (unsigned)((ga >> 32) & 0x01FFFFFFull) | (2u << 30); // [56:32]+type=2
  v8i g1;
  g1[0] = (int)(1u << 16);                       // data_size=1 (2 bytes)
  g1[1] = (int)((dim0 & 0xFFFFu) << 16);         // tensor_dim0[15:0] @ bit48
  g1[2] = (int)(((dim0 >> 16) & 0xFFFFu) | ((dim1 & 0xFFFFu) << 16));
  g1[3] = (int)(((dim1 >> 16) & 0xFFFFu) | ((dim0 & 0xFFFFu) << 16)); // tile_dim0
  g1[4] = (int)(dim1 & 0xFFFFu);                 // tile_dim1 (tile_dim2=0)
  g1[5] = (int)stride0;                          // tensor_dim0_stride[31:0]
  g1[6] = 0;
  g1[7] = 0;
  v4i z4 = (v4i){0, 0, 0, 0};
#if __clang_major__ >= 23
  v8i z8 = (v8i){0, 0, 0, 0, 0, 0, 0, 0};
  __builtin_amdgcn_tensor_load_to_lds(g0, g1, z4, z4, z8, 0);
#else
  __builtin_amdgcn_tensor_load_to_lds(g0, g1, z4, z4, 0);
#endif
}

// ---------------------------------------------------------------- converts --
__global__ __launch_bounds__(256) void cvt_bf16_kernel(
    const float* __restrict__ src, unsigned short* __restrict__ dst, int n) {
  int i = blockIdx.x * blockDim.x + threadIdx.x;
  if (i < n) dst[i] = f2bf(src[i]);
}

// Pack a (HID x KCOMB) fp32 weight matrix's columns [colOff, colOff+1024) into
// WMMA B-fragment layout: tile (nt, kc); lane holds 16 contiguous bf16:
//   lane 0..15  -> column nt*16+lane,      K = kc*32 + 0..15
//   lane 16..31 -> column nt*16+(lane-16), K = kc*32 + 16..31
__global__ __launch_bounds__(256) void pack_w_kernel(
    const float* __restrict__ W, unsigned short* __restrict__ dst, int colOff) {
  int tid  = blockIdx.x * blockDim.x + threadIdx.x;  // 65536 = 64*32*32
  int lane = tid & 31;
  int kc   = (tid >> 5) & 31;
  int nt   = tid >> 10;                              // 0..63
  int n    = nt * 16 + (lane & 15);
  int kb   = kc * 32 + (lane >> 4) * 16;
  const float* src = W + (size_t)n * KCOMB + colOff + kb;
  unsigned short* d = dst + (size_t)tid * 16;
  #pragma unroll
  for (int j = 0; j < 16; ++j) d[j] = f2bf(src[j]);
}

// ----------------------------------------------------- big input-proj GEMM --
// Zx[g][row][n] = sum_k x[row][k] * W_g[n][k]  (bf16 out), 4Mx2N per wave.
__global__ __launch_bounds__(256) void gemm_x_kernel(
    const unsigned short* __restrict__ xbf,
    const unsigned short* __restrict__ wxpk,
    unsigned short* __restrict__ zx) {
  int gtid = blockIdx.x * blockDim.x + threadIdx.x;
  int wave = gtid >> 5;                 // 0 .. 32767
  int lane = threadIdx.x & 31;
  int mt4  = wave & 255;                // group of 4 row tiles
  int nt2  = (wave >> 8) & 31;          // group of 2 col tiles
  int g    = wave >> 13;                // gate 0..3
  int half = lane >> 4;
  int m    = lane & 15;
  int m0   = mt4 * 64;

  const unsigned short* arow = xbf + (size_t)(m0 + m) * DIN + half * 16;
  const unsigned short* bp0 =
      wxpk + (size_t)((g * 64 + nt2 * 2) * 32) * 512 + lane * 16;
  const unsigned short* bp1 = bp0 + 32 * 512;

  v8f acc[4][2];
  #pragma unroll
  for (int i = 0; i < 4; ++i)
    #pragma unroll
    for (int j = 0; j < 2; ++j)
      acc[i][j] = (v8f){0.f, 0.f, 0.f, 0.f, 0.f, 0.f, 0.f, 0.f};

  #pragma unroll 1
  for (int kc = 0; kc < 32; ++kc) {
    FragB16 a[4], b[2];
    #pragma unroll
    for (int i = 0; i < 4; ++i) {
      const unsigned short* ap = arow + (size_t)i * 16 * DIN + kc * 32;
      a[i].q[0] = *(const uint4*)(ap);
      a[i].q[1] = *(const uint4*)(ap + 8);
    }
    b[0].q[0] = *(const uint4*)(bp0 + (size_t)kc * 512);
    b[0].q[1] = *(const uint4*)(bp0 + (size_t)kc * 512 + 8);
    b[1].q[0] = *(const uint4*)(bp1 + (size_t)kc * 512);
    b[1].q[1] = *(const uint4*)(bp1 + (size_t)kc * 512 + 8);
    #pragma unroll
    for (int i = 0; i < 4; ++i)
      #pragma unroll
      for (int j = 0; j < 2; ++j)
        acc[i][j] = __builtin_amdgcn_wmma_f32_16x16x32_bf16(
            false, a[i].v, false, b[j].v, (short)0, acc[i][j], false, false);
  }
  unsigned short* zg = zx + (size_t)g * ((size_t)ROWS * HID);
  #pragma unroll
  for (int i = 0; i < 4; ++i)
    #pragma unroll
    for (int j = 0; j < 2; ++j)
      #pragma unroll
      for (int r = 0; r < 8; ++r) {
        int row = m0 + i * 16 + r + half * 8;
        int col = (nt2 * 2 + j) * 16 + m;
        zg[(size_t)row * HID + col] = f2bf(acc[i][j][r]);
      }
}

// ------------------------------------------------------ per-step recurrent --
// zh[b][g*1024+n] = sum_k h[b][k] * W_g[n][D+k]  (fp32 out).
// h (64x1024 bf16, 128KB) is staged into LDS once per block by the TDM.
__global__ __launch_bounds__(256) void gemm_h_kernel(
    const unsigned short* __restrict__ hbf,
    const unsigned short* __restrict__ whpk,
    float* __restrict__ zh) {
  __shared__ unsigned short smem[BATCH * HID];   // 128 KB of the 320KB WGP LDS
  int bid  = blockIdx.x;                 // 0..127
  int g    = bid >> 5;                   // gate 0..3
  int ntb  = (bid & 31) * 2;             // first of 2 col tiles for this block
  int wv   = threadIdx.x >> 5;           // wave in block 0..7
  int lane = threadIdx.x & 31;
  int mt   = wv & 3;                     // batch tile 0..3
  int nt   = ntb + (wv >> 2);            // col tile
  int half = lane >> 4;
  int m    = lane & 15;

  if (threadIdx.x < 32) {                // one wave drives the DMA
    tdm_load_2d_bf16_to_lds(hbf, (unsigned)(uintptr_t)&smem[0],
                            HID, BATCH, HID);
    __builtin_amdgcn_s_wait_tensorcnt(0);
  }
  __syncthreads();

  const unsigned short* as = &smem[0] + (size_t)(mt * 16 + m) * HID + half * 16;
  const unsigned short* bptr =
      whpk + (size_t)((g * 64 + nt) * 32) * 512 + lane * 16;

  v8f acc = (v8f){0.f, 0.f, 0.f, 0.f, 0.f, 0.f, 0.f, 0.f};
  #pragma unroll 4
  for (int kc = 0; kc < 32; ++kc) {
    FragB16 a, b;
    a.q[0] = *(const uint4*)(as + kc * 32);        // ds_load_b128
    a.q[1] = *(const uint4*)(as + kc * 32 + 8);
    b.q[0] = *(const uint4*)(bptr + (size_t)kc * 512);
    b.q[1] = *(const uint4*)(bptr + (size_t)kc * 512 + 8);
    acc = __builtin_amdgcn_wmma_f32_16x16x32_bf16(
        false, a.v, false, b.v, (short)0, acc, false, false);
  }
  #pragma unroll
  for (int r = 0; r < 8; ++r) {
    int row = mt * 16 + r + half * 8;
    zh[(size_t)row * (4 * HID) + g * HID + nt * 16 + m] = acc[r];
  }
}

// ------------------------------------------------------- fused cell update --
__global__ __launch_bounds__(256) void lstm_cell_kernel(
    const unsigned short* __restrict__ zx, const float* __restrict__ zh,
    const float* __restrict__ bF, const float* __restrict__ gbF,
    const float* __restrict__ bI, const float* __restrict__ gbI,
    const float* __restrict__ bU, const float* __restrict__ gbU,
    const float* __restrict__ bO, const float* __restrict__ gbO,
    float* __restrict__ cst, unsigned short* __restrict__ hbf,
    float* __restrict__ out, int t) {
  int idx = blockIdx.x * blockDim.x + threadIdx.x;   // 0..65535
  int b = idx >> 10;
  int h = idx & 1023;
  const size_t GSZ = (size_t)ROWS * HID;
  size_t zrow = ((size_t)t * BATCH + b) * HID + h;
  size_t hrow = (size_t)b * (4 * HID) + h;

  float zf = bf2f(zx[zrow])           + zh[hrow]        + bF[h];
  float zi = bf2f(zx[GSZ + zrow])     + zh[hrow + 1024] + bI[h];
  float zu = bf2f(zx[2 * GSZ + zrow]) + zh[hrow + 2048] + bU[h];
  float zo = bf2f(zx[3 * GSZ + zrow]) + zh[hrow + 3072] + bO[h];

  float f = sigmoidf_(__cosf(zf) + gbF[h]);
  float i = sigmoidf_(__cosf(zi) + gbI[h]);
  float u = tanhf(__cosf(zu) + gbU[h]);
  float o = sigmoidf_(__cosf(zo) + gbO[h]);

  float cn = f * cst[idx] + i * u;
  float hn = o * tanhf(cn);
  cst[idx] = cn;
  hbf[idx] = f2bf(hn);
  out[(size_t)t * BH + idx] = hn;
}

__global__ __launch_bounds__(256) void init_state_kernel(
    float* __restrict__ c, unsigned short* __restrict__ hbf) {
  int idx = blockIdx.x * blockDim.x + threadIdx.x;
  if (idx < BH) { c[idx] = 0.0f; hbf[idx] = 0; }
}

__global__ __launch_bounds__(256) void tail_kernel(
    float* __restrict__ out, const float* __restrict__ c) {
  int idx = blockIdx.x * blockDim.x + threadIdx.x;   // 0 .. 2*BH-1
  if (idx < BH)
    out[(size_t)TSEQ * BH + idx] = out[(size_t)(TSEQ - 1) * BH + idx]; // hx
  else
    out[(size_t)TSEQ * BH + idx] = c[idx - BH];                        // cx
}

// -------------------------------------------------------------------- host --
extern "C" void kernel_launch(void* const* d_in, const int* in_sizes, int n_in,
                              void* d_out, int out_size, void* d_ws, size_t ws_size,
                              hipStream_t stream) {
  const float* x   = (const float*)d_in[0];
  const float* Wg[4] = { (const float*)d_in[1], (const float*)d_in[4],
                         (const float*)d_in[7], (const float*)d_in[10] };
  const float* bF  = (const float*)d_in[2];
  const float* gbF = (const float*)d_in[3];
  const float* bI  = (const float*)d_in[5];
  const float* gbI = (const float*)d_in[6];
  const float* bU  = (const float*)d_in[8];
  const float* gbU = (const float*)d_in[9];
  const float* bO  = (const float*)d_in[11];
  const float* gbO = (const float*)d_in[12];
  float* out = (float*)d_out;

  char* ws = (char*)d_ws;
  size_t off = 0;
  auto carve = [&](size_t bytes) -> char* {
    char* p = ws + off;
    off = (off + bytes + 255) & ~(size_t)255;
    return p;
  };
  const size_t GSZ = (size_t)ROWS * HID;
  unsigned short* zx   = (unsigned short*)carve(4 * GSZ * sizeof(unsigned short));
  unsigned short* xbf  = (unsigned short*)carve((size_t)ROWS * DIN * sizeof(unsigned short));
  unsigned short* wxpk = (unsigned short*)carve((size_t)4 * 64 * 32 * 32 * 16 * sizeof(unsigned short));
  unsigned short* whpk = (unsigned short*)carve((size_t)4 * 64 * 32 * 32 * 16 * sizeof(unsigned short));
  float*          zh   = (float*)carve((size_t)BATCH * 4 * HID * sizeof(float));
  float*          cst  = (float*)carve((size_t)BH * sizeof(float));
  unsigned short* hbf  = (unsigned short*)carve((size_t)BH * sizeof(unsigned short));
  (void)ws_size; (void)n_in; (void)in_sizes; (void)out_size;

  // 1) convert inputs to bf16
  {
    int n = ROWS * DIN;                                  // 16,777,216
    cvt_bf16_kernel<<<n / 256, 256, 0, stream>>>(x, xbf, n);
  }
  // 2) pack weights (x-part and h-part) into WMMA B-fragment layout
  for (int g = 0; g < 4; ++g) {
    pack_w_kernel<<<256, 256, 0, stream>>>(Wg[g], wxpk + (size_t)g * (64 * 32 * 32 * 16), 0);
    pack_w_kernel<<<256, 256, 0, stream>>>(Wg[g], whpk + (size_t)g * (64 * 32 * 32 * 16), DIN);
  }
  // 3) big parallel input-projection GEMM (all T, all gates), 4Mx2N blocked
  gemm_x_kernel<<<4096, 256, 0, stream>>>(xbf, wxpk, zx);
  // 4) init recurrent state
  init_state_kernel<<<BH / 256, 256, 0, stream>>>(cst, hbf);
  // 5) sequential scan: TDM-staged recurrent GEMM + fused cell update
  for (int t = 0; t < TSEQ; ++t) {
    gemm_h_kernel<<<128, 256, 0, stream>>>(hbf, whpk, zh);
    lstm_cell_kernel<<<BH / 256, 256, 0, stream>>>(
        zx, zh, bF, gbF, bI, gbI, bU, gbU, bO, gbO, cst, hbf, out, t);
  }
  // 6) append (hx, cx)
  tail_kernel<<<(2 * BH) / 256, 256, 0, stream>>>(out, cst);
}
